// DatabricksExperts_89833535963319
// MI455X (gfx1250) — compile-verified, matrix operand order checked by reading, not run on hardware
//
#include <hip/hip_runtime.h>

typedef __bf16 bf16_t;
typedef __attribute__((ext_vector_type(16))) __bf16 v16bf;
typedef __attribute__((ext_vector_type(8)))  __bf16 v8bf;
typedef __attribute__((ext_vector_type(2)))  __bf16 v2bf;
typedef __attribute__((ext_vector_type(8)))  float  v8f;

namespace {
constexpr int kT = 2048;   // B*S tokens
constexpr int kD = 2048;   // model dim
constexpr int kF = 4096;   // ffn dim
constexpr int kE = 8;      // experts

constexpr int MB = 64;     // token rows per block
constexpr int NB = 128;    // output cols per block
constexpr int KB = 32;     // staged K chunk (1 WMMA k-step), double buffered
constexpr int LROW = KB + 8;  // padded LDS row in bf16 elems (80B stride)

// workspace layout (bytes)
constexpr size_t kOffTopW = 0;          // T*2 f32
constexpr size_t kOffTopE = 16384;      // T*2 i32
constexpr size_t kOffCnt  = 32768;      // E   i32
constexpr size_t kOffList = 36864;      // E*T i32
constexpr size_t kOffG    = 131072;     // 2*T*F bf16 (32 MB)
}

__device__ __forceinline__ v16bf load_frag(const bf16_t* p) {
  // 16-bit WMMA operand layout: lane holds K {0..7} at p, K {16..23} at p+16
  // (lanes 16..31 use the +8 shifted halves via the caller's khalf offset).
  v8bf lo = *(const v8bf*)p;
  v8bf hi = *(const v8bf*)(p + 16);
  return __builtin_shufflevector(lo, hi, 0,1,2,3,4,5,6,7,8,9,10,11,12,13,14,15);
}

__device__ __forceinline__ void async_copy_16B_to_lds(const bf16_t* gsrc, bf16_t* lds_dst) {
  // GLOBAL_LOAD_ASYNC_TO_LDS_B128: per-lane 16B memory->LDS, tracked by ASYNCcnt.
  // LDS aperture keeps the byte offset in addr[31:0].
  const unsigned lds_off = (unsigned)(unsigned long long)lds_dst;
  const unsigned long long gaddr = (unsigned long long)gsrc;
  asm volatile("global_load_async_to_lds_b128 %0, %1, off"
               :: "v"(lds_off), "v"(gaddr) : "memory");
}

// ---------------------------------------------------------------------------
// Kernel 1: router — logits, softmax, top-2, renorm, per-expert token lists
// ---------------------------------------------------------------------------
__global__ __launch_bounds__(256) void moe_router_kernel(
    const float* __restrict__ x, const float* __restrict__ rw,
    float* __restrict__ topw, int* __restrict__ tope,
    int* __restrict__ cnt, int* __restrict__ list)
{
  const int wave = threadIdx.x >> 5;
  const int lane = threadIdx.x & 31;
  const int t = blockIdx.x * 8 + wave;
  if (t >= kT) return;

  float acc[kE];
#pragma unroll
  for (int e = 0; e < kE; ++e) acc[e] = 0.0f;

  const float* xr = x + (size_t)t * kD;
  for (int d = lane; d < kD; d += 32) {
    const float xv = xr[d];
    const float* r = rw + (size_t)d * kE;   // router_w [D, E], E contiguous
#pragma unroll
    for (int e = 0; e < kE; ++e) acc[e] += xv * r[e];
  }
#pragma unroll
  for (int off = 16; off > 0; off >>= 1) {
#pragma unroll
    for (int e = 0; e < kE; ++e) acc[e] += __shfl_xor(acc[e], off, 32);
  }

  if (lane == 0) {
    float m = acc[0];
#pragma unroll
    for (int e = 1; e < kE; ++e) m = fmaxf(m, acc[e]);
    float p[kE];
#pragma unroll
    for (int e = 0; e < kE; ++e) p[e] = __expf(acc[e] - m);
    int i0 = 0;
#pragma unroll
    for (int e = 1; e < kE; ++e) if (p[e] > p[i0]) i0 = e;   // first max (tie: low idx)
    int i1 = (i0 == 0) ? 1 : 0;
#pragma unroll
    for (int e = 0; e < kE; ++e) if (e != i0 && p[e] > p[i1]) i1 = e;
    const float s2 = p[i0] + p[i1];        // softmax denom cancels in renorm
    topw[t * 2 + 0] = p[i0] / s2;
    topw[t * 2 + 1] = p[i1] / s2;
    tope[t * 2 + 0] = i0;
    tope[t * 2 + 1] = i1;
    const int s0 = atomicAdd(&cnt[i0], 1);
    list[i0 * kT + s0] = t * 2 + 0;        // encode (token, k-slot)
    const int s1 = atomicAdd(&cnt[i1], 1);
    list[i1 * kT + s1] = t * 2 + 1;
  }
}

// ---------------------------------------------------------------------------
// Kernel 2: per-expert gated up-projection  G[row] = silu(x@w1_e) * (x@v1_e)
// grid: (T/MB m-tiles FASTEST, F/NB, E): m-tile blocks sharing a weight slice
// are adjacent in dispatch order -> weight slice streams HBM once, siblings
// hit L2. Double-buffered LDS + register-pipelined global loads + prefetch.
// ---------------------------------------------------------------------------
__global__ __launch_bounds__(256) void moe_up_kernel(
    const float* __restrict__ x,
    const float* __restrict__ w1, const float* __restrict__ v1,
    const int* __restrict__ cnt, const int* __restrict__ list,
    bf16_t* __restrict__ G)
{
  __shared__ bf16_t Xs [2 * MB * LROW];
  __shared__ bf16_t W1t[2 * NB * LROW];   // transposed: [f_local][k]
  __shared__ bf16_t V1t[2 * NB * LROW];
  __shared__ int sRow[MB];
  __shared__ int sTok[MB];

  const int e = blockIdx.z;
  const int nTok = cnt[e];
  const int mbase = blockIdx.x * MB;      // m-tile: fastest grid dim
  if (mbase >= nTok) return;
  const int f0 = blockIdx.y * NB;

  const int tid = threadIdx.x;
  if (tid < MB) {
    int row = -1, tok = 0;
    const int idx = mbase + tid;
    if (idx < nTok) {
      const int entry = list[e * kT + idx];
      const int t = entry >> 1;
      if ((unsigned)t < (unsigned)kT) { tok = t; row = (entry & 1) * kT + t; }
    }
    sTok[tid] = tok;
    sRow[tid] = row;
  }
  __syncthreads();

  const int wave = tid >> 5;
  const int lane = tid & 31;
  const int mw   = wave >> 1;           // 0..3 : m-tile of 16
  const int nw0  = (wave & 1) * 4;      // 0/4  : first of 4 n-tiles
  const int frow = lane & 15;
  const int khalf = (lane >> 4) * 8;

  v8f ch[4], cv[4];
#pragma unroll
  for (int j = 0; j < 4; ++j)
#pragma unroll
    for (int i = 0; i < 8; ++i) { ch[j][i] = 0.0f; cv[j][i] = 0.0f; }

  // staging thread maps
  const int xr = tid >> 2;              // 0..63 token row
  const int xc = (tid & 3) * 8;         // 8 f32 per thread (two float4)
  const size_t xrow = (size_t)sTok[xr] * kD;
  const int fq = (tid & 31) * 4;        // float4 group along F
  const int kr = tid >> 5;              // 0..7 base k-row (4 rows per thread)
  const float* w1e = w1 + (size_t)e * kD * kF;
  const float* v1e = v1 + (size_t)e * kD * kF;

  float4 rx0, rx1, rw4[4], rv4[4];

  auto do_load = [&](int d0) {
    const float* xp = x + xrow + d0 + xc;
    rx0 = *(const float4*)xp;
    rx1 = *(const float4*)(xp + 4);
#pragma unroll
    for (int s = 0; s < 4; ++s) {
      const size_t g = (size_t)(d0 + kr + 8 * s) * kF + f0 + fq;
      rw4[s] = *(const float4*)(w1e + g);
      rv4[s] = *(const float4*)(v1e + g);
    }
  };

  auto do_store = [&](int buf) {
    bf16_t* Xb = Xs  + buf * MB * LROW;
    bf16_t* Wb = W1t + buf * NB * LROW;
    bf16_t* Vb = V1t + buf * NB * LROW;
    v8bf px;
    px[0] = (bf16_t)rx0.x; px[1] = (bf16_t)rx0.y; px[2] = (bf16_t)rx0.z; px[3] = (bf16_t)rx0.w;
    px[4] = (bf16_t)rx1.x; px[5] = (bf16_t)rx1.y; px[6] = (bf16_t)rx1.z; px[7] = (bf16_t)rx1.w;
    *(v8bf*)&Xb[xr * LROW + xc] = px;   // single ds_store_b128
#pragma unroll
    for (int s = 0; s < 4; ++s) {
      const int d = kr + 8 * s;
      Wb[(fq + 0) * LROW + d] = (bf16_t)rw4[s].x;
      Wb[(fq + 1) * LROW + d] = (bf16_t)rw4[s].y;
      Wb[(fq + 2) * LROW + d] = (bf16_t)rw4[s].z;
      Wb[(fq + 3) * LROW + d] = (bf16_t)rw4[s].w;
      Vb[(fq + 0) * LROW + d] = (bf16_t)rv4[s].x;
      Vb[(fq + 1) * LROW + d] = (bf16_t)rv4[s].y;
      Vb[(fq + 2) * LROW + d] = (bf16_t)rv4[s].z;
      Vb[(fq + 3) * LROW + d] = (bf16_t)rv4[s].w;
    }
  };

  auto do_compute = [&](int buf) {
    const bf16_t* Xb = Xs  + buf * MB * LROW;
    const bf16_t* Wb = W1t + buf * NB * LROW;
    const bf16_t* Vb = V1t + buf * NB * LROW;
    const v16bf a = load_frag(&Xb[(mw * 16 + frow) * LROW + khalf]);
    v16bf bw[4], bv[4];
#pragma unroll
    for (int j = 0; j < 4; ++j) {
      bw[j] = load_frag(&Wb[((nw0 + j) * 16 + frow) * LROW + khalf]);
      bv[j] = load_frag(&Vb[((nw0 + j) * 16 + frow) * LROW + khalf]);
    }
#pragma unroll
    for (int j = 0; j < 4; ++j) {
      ch[j] = __builtin_amdgcn_wmma_f32_16x16x32_bf16(false, a, false, bw[j],
                                                      (short)0, ch[j], false, false);
      cv[j] = __builtin_amdgcn_wmma_f32_16x16x32_bf16(false, a, false, bv[j],
                                                      (short)0, cv[j], false, false);
    }
  };

  do_load(0);
  int buf = 0;
  for (int d0 = 0; d0 < kD; d0 += KB, buf ^= 1) {
    do_store(buf);
    __syncthreads();
    if (d0 + KB < kD) do_load(d0 + KB);          // consumed next iter: waits land after WMMAs
    if (d0 + 2 * KB < kD) {                       // warm L2 one chunk further
      __builtin_prefetch(w1e + (size_t)(d0 + 2 * KB + kr) * kF + f0 + fq, 0, 0);
      __builtin_prefetch(v1e + (size_t)(d0 + 2 * KB + kr) * kF + f0 + fq, 0, 0);
    }
    do_compute(buf);
  }

  // epilogue: g = silu(h) * v  -> bf16 scatter rows into G
  const int mtop = (lane >> 4) * 8;   // C layout: lanes 16..31 hold M=8..15
#pragma unroll
  for (int j = 0; j < 4; ++j) {
#pragma unroll
    for (int i = 0; i < 8; ++i) {
      const int mloc = mw * 16 + mtop + i;
      const int row = sRow[mloc];
      if (row >= 0) {
        const float h = ch[j][i];
        const float g = (h / (1.0f + __expf(-h))) * cv[j][i];
        G[(size_t)row * kF + f0 + (nw0 + j) * 16 + frow] = (bf16_t)g;
      }
    }
  }
}

// ---------------------------------------------------------------------------
// Kernel 3: per-expert down projection  out[t] += w_{t,e} * (G_row @ w2_e)
// grid: (T/MB m-tiles FASTEST, D/NB, E) for w2 L2 reuse across m-tiles.
// G tile staged via async global->LDS (ASYNCcnt), pipelined one chunk ahead.
// ---------------------------------------------------------------------------
__global__ __launch_bounds__(256) void moe_down_kernel(
    const bf16_t* __restrict__ G, const float* __restrict__ w2,
    const int* __restrict__ cnt, const int* __restrict__ list,
    const float* __restrict__ topw,
    float* __restrict__ out)
{
  __shared__ bf16_t Gs [2 * MB * LROW];
  __shared__ bf16_t W2t[2 * NB * LROW];   // transposed: [d_local][k]
  __shared__ int   sTok[MB];
  __shared__ int   sGRow[MB];
  __shared__ float sWgt[MB];

  const int e = blockIdx.z;
  const int nTok = cnt[e];
  const int mbase = blockIdx.x * MB;      // m-tile: fastest grid dim
  if (mbase >= nTok) return;
  const int dd0 = blockIdx.y * NB;

  const int tid = threadIdx.x;
  if (tid < MB) {
    int tok = -1, grow = 0; float wgt = 0.0f;
    const int idx = mbase + tid;
    if (idx < nTok) {
      const int entry = list[e * kT + idx];
      const int t = entry >> 1, k = entry & 1;
      if ((unsigned)t < (unsigned)kT) { tok = t; grow = k * kT + t; wgt = topw[t * 2 + k]; }
    }
    sTok[tid] = tok; sGRow[tid] = grow; sWgt[tid] = wgt;
  }
  __syncthreads();

  const int wave = tid >> 5;
  const int lane = tid & 31;
  const int mw   = wave >> 1;
  const int nw0  = (wave & 1) * 4;
  const int frow = lane & 15;
  const int khalf = (lane >> 4) * 8;

  v8f c[4];
#pragma unroll
  for (int j = 0; j < 4; ++j)
#pragma unroll
    for (int i = 0; i < 8; ++i) c[j][i] = 0.0f;

  const int gr = tid >> 2;
  const int gc = (tid & 3) * 8;          // 8 bf16 = 16B per thread
  const size_t growoff = (size_t)sGRow[gr] * kF;
  const int fq = (tid & 31) * 4;
  const int kr = tid >> 5;
  const float* w2e = w2 + (size_t)e * kF * kD;

  float4 rw4[4];

  auto g_async = [&](int k0, int bb) {
    async_copy_16B_to_lds(G + growoff + k0 + gc, &Gs[bb * MB * LROW + gr * LROW + gc]);
  };
  auto w_load = [&](int k0) {
#pragma unroll
    for (int s = 0; s < 4; ++s)
      rw4[s] = *(const float4*)(w2e + (size_t)(k0 + kr + 8 * s) * kD + dd0 + fq);
  };
  auto w_store = [&](int bb) {
    bf16_t* Wb = W2t + bb * NB * LROW;
#pragma unroll
    for (int s = 0; s < 4; ++s) {
      const int d = kr + 8 * s;
      Wb[(fq + 0) * LROW + d] = (bf16_t)rw4[s].x;
      Wb[(fq + 1) * LROW + d] = (bf16_t)rw4[s].y;
      Wb[(fq + 2) * LROW + d] = (bf16_t)rw4[s].z;
      Wb[(fq + 3) * LROW + d] = (bf16_t)rw4[s].w;
    }
  };
  auto do_compute = [&](int bb) {
    const bf16_t* Gb = Gs  + bb * MB * LROW;
    const bf16_t* Wb = W2t + bb * NB * LROW;
    const v16bf a = load_frag(&Gb[(mw * 16 + frow) * LROW + khalf]);
    v16bf b[4];
#pragma unroll
    for (int j = 0; j < 4; ++j)
      b[j] = load_frag(&Wb[((nw0 + j) * 16 + frow) * LROW + khalf]);
#pragma unroll
    for (int j = 0; j < 4; ++j)
      c[j] = __builtin_amdgcn_wmma_f32_16x16x32_bf16(false, a, false, b[j],
                                                     (short)0, c[j], false, false);
  };

  g_async(0, 0);
  w_load(0);
  int buf = 0;
  for (int k0 = 0; k0 < kF; k0 += KB, buf ^= 1) {
    w_store(buf);
    if (k0 + KB < kF) {
      g_async(k0 + KB, buf ^ 1);                   // 1 outstanding for next chunk
      asm volatile("s_wait_asynccnt 0x1" ::: "memory");  // oldest (this chunk) done
    } else {
      asm volatile("s_wait_asynccnt 0x0" ::: "memory");
    }
    __syncthreads();
    if (k0 + KB < kF) w_load(k0 + KB);             // waits land after WMMAs
    if (k0 + 2 * KB < kF)
      __builtin_prefetch(w2e + (size_t)(k0 + 2 * KB + kr) * kD + dd0 + fq, 0, 0);
    do_compute(buf);
  }

  // epilogue: scale by routing weight; exactly 2 commutative atomic adds per out elem
  const int mtop = (lane >> 4) * 8;
#pragma unroll
  for (int j = 0; j < 4; ++j) {
#pragma unroll
    for (int i = 0; i < 8; ++i) {
      const int mloc = mw * 16 + mtop + i;
      const int t = sTok[mloc];
      if (t >= 0) {
        atomicAdd(&out[(size_t)t * kD + dd0 + (nw0 + j) * 16 + frow],
                  c[j][i] * sWgt[mloc]);
      }
    }
  }
}

// ---------------------------------------------------------------------------
extern "C" void kernel_launch(void* const* d_in, const int* in_sizes, int n_in,
                              void* d_out, int out_size, void* d_ws, size_t ws_size,
                              hipStream_t stream) {
  (void)in_sizes; (void)n_in; (void)out_size; (void)ws_size;
  const float* x  = (const float*)d_in[0];   // [T, D]
  const float* rw = (const float*)d_in[1];   // [D, E]
  const float* w1 = (const float*)d_in[2];   // [E, D, F]
  const float* v1 = (const float*)d_in[3];   // [E, D, F]
  const float* w2 = (const float*)d_in[4];   // [E, F, D]
  float* out = (float*)d_out;

  char* ws = (char*)d_ws;
  float*  topw = (float*)(ws + kOffTopW);
  int*    tope = (int*)  (ws + kOffTopE);
  int*    cnt  = (int*)  (ws + kOffCnt);
  int*    list = (int*)  (ws + kOffList);
  bf16_t* G    = (bf16_t*)(ws + kOffG);

  hipMemsetAsync(cnt, 0, kE * sizeof(int), stream);
  hipMemsetAsync(out, 0, (size_t)kT * kD * sizeof(float), stream);

  moe_router_kernel<<<kT / 8, 256, 0, stream>>>(x, rw, topw, tope, cnt, list);
  // m-tiles fastest so blocks sharing a weight slice are dispatch-adjacent (L2 reuse)
  moe_up_kernel<<<dim3(kT / MB, kF / NB, kE), 256, 0, stream>>>(x, w1, v1, cnt, list, G);
  moe_down_kernel<<<dim3(kT / MB, kD / NB, kE), 256, 0, stream>>>(G, w2, cnt, list, topw, out);
}
